// my_transformer_56401510531551
// MI455X (gfx1250) — matrix operand order repaired and checked by reading, not compiled, not run on hardware
//
#include <hip/hip_runtime.h>

// ---------------------------------------------------------------------------
// Problem constants (from the reference)
// ---------------------------------------------------------------------------
#define BB   8
#define SS   1000
#define DD   1024
#define KK   1024
#define HH   16
#define HDIM 64
#define MIDD 4096
#define OUTN 1000

typedef __attribute__((ext_vector_type(16))) __bf16 bf16x16;
typedef __attribute__((ext_vector_type(8)))  float  f32x8;

union FragB {
    bf16x16 v;
    uint4   q[2];
    __bf16  e[16];
};

__device__ __forceinline__ f32x8 wmma_bf16(const bf16x16& a, const bf16x16& b, f32x8 c) {
    return __builtin_amdgcn_wmma_f32_16x16x32_bf16(
        /*neg_a=*/false, a, /*neg_b=*/false, b,
        /*c_mod=*/(short)0, c, /*reuse_a=*/false, /*reuse_b=*/false);
}

// ---------------------------------------------------------------------------
// fp32 -> bf16 convert
// ---------------------------------------------------------------------------
__global__ void xf_f2bf(const float* __restrict__ src, __bf16* __restrict__ dst, size_t n) {
    size_t i = (size_t)blockIdx.x * blockDim.x + threadIdx.x;
    if (i < n) dst[i] = (__bf16)src[i];
}

// ---------------------------------------------------------------------------
// x + positional encoding -> bf16
// ---------------------------------------------------------------------------
__global__ void xf_posenc(const float* __restrict__ x, __bf16* __restrict__ xpe) {
    size_t i = (size_t)blockIdx.x * blockDim.x + threadIdx.x;
    size_t tot = (size_t)BB * SS * DD;
    if (i >= tot) return;
    int d = (int)(i % DD);
    int s = (int)((i / DD) % SS);
    int j2 = (d >> 1) * 2;
    const float LOG1E4 = 9.210340371976184f;
    float div = __expf(-(LOG1E4 / (float)DD) * (float)j2);
    float ang = (float)s * div;
    float pe  = (d & 1) ? __cosf(ang) : __sinf(ang);
    xpe[i] = (__bf16)(x[i] + pe);
}

// ---------------------------------------------------------------------------
// Tiled bf16 WMMA GEMM:  C[M,N] = A[M,Kd] * B[Kd,N]  (+ epilogue)
// Block 256 thr (8 waves).  Macro tile 128x128, Kstep 32.
// Wave (wm,wn) in 2x4 grid computes 64x32 = 4x2 tiles of 16x16.
// FLAGS: 1=store bf16, 2=store f32, 4=+bias, 8=relu, 16=+residual(f32)
// ---------------------------------------------------------------------------
#define LDA_S 48   // padded LDS row stride (elems): 96B rows, 16B aligned chunks

template <int FLAGS>
__global__ __launch_bounds__(256)
void xf_gemm(const __bf16* __restrict__ A, const __bf16* __restrict__ Bm,
             int M, int N, int Kd,
             const float* __restrict__ bias, const float* __restrict__ resid,
             __bf16* __restrict__ outb, float* __restrict__ outf) {
    __shared__ __bf16 As[128 * LDA_S];
    __shared__ __bf16 Bs[128 * LDA_S];   // stored transposed: Bs[n][k]

    const int t     = threadIdx.x;
    const int lane  = t & 31;
    const int wv    = t >> 5;
    const int wm    = wv & 1;       // 0..1  (M dir, 64 rows each)
    const int wn    = wv >> 1;      // 0..3  (N dir, 32 cols each)
    const int lmod  = lane & 15;
    const int lhalf = lane >> 4;

    const int m0 = blockIdx.y * 128;
    const int n0 = blockIdx.x * 128;

    // staging coords
    const int arow = t >> 1;           // 0..127
    const int acol = (t & 1) * 16;     // 0 / 16
    const int brow = t >> 3;           // k 0..31
    const int bcol = (t & 7) * 16;     // n seg

    int gm = m0 + arow; if (gm >= M) gm = M - 1;

    f32x8 acc[4][2];
    for (int i = 0; i < 4; ++i)
        for (int j = 0; j < 2; ++j)
            acc[i][j] = (f32x8){0.f,0.f,0.f,0.f,0.f,0.f,0.f,0.f};

    for (int k0 = 0; k0 < Kd; k0 += 32) {
        // ---- stage A tile (row-major) ----
        {
            const uint4* src = (const uint4*)(A + (size_t)gm * Kd + k0 + acol);
            uint4 a0 = src[0], a1 = src[1];
            *(uint4*)(As + arow * LDA_S + acol)     = a0;
            *(uint4*)(As + arow * LDA_S + acol + 8) = a1;
        }
        // ---- stage B tile transposed ----
        {
            const uint4* src = (const uint4*)(Bm + (size_t)(k0 + brow) * N + n0 + bcol);
            uint4 b0 = src[0], b1 = src[1];
            const __bf16* bp = (const __bf16*)&b0;
            #pragma unroll
            for (int i = 0; i < 8; ++i) Bs[(bcol + i) * LDA_S + brow] = bp[i];
            bp = (const __bf16*)&b1;
            #pragma unroll
            for (int i = 0; i < 8; ++i) Bs[(bcol + 8 + i) * LDA_S + brow] = bp[i];
            if (k0 + 32 < Kd)
                __builtin_prefetch(Bm + (size_t)(k0 + 32 + brow) * N + n0 + bcol, 0, 1);
        }
        __syncthreads();

        // ---- fragments ----
        FragB afr[4], bfr[2];
        #pragma unroll
        for (int tm = 0; tm < 4; ++tm) {
            const __bf16* ap = As + (wm * 64 + tm * 16 + lmod) * LDA_S + lhalf * 8;
            afr[tm].q[0] = *(const uint4*)(ap);
            afr[tm].q[1] = *(const uint4*)(ap + 16);
        }
        #pragma unroll
        for (int tn = 0; tn < 2; ++tn) {
            const __bf16* bp = Bs + (wn * 32 + tn * 16 + lmod) * LDA_S + lhalf * 8;
            bfr[tn].q[0] = *(const uint4*)(bp);
            bfr[tn].q[1] = *(const uint4*)(bp + 16);
        }
        #pragma unroll
        for (int tm = 0; tm < 4; ++tm)
            #pragma unroll
            for (int tn = 0; tn < 2; ++tn)
                acc[tm][tn] = wmma_bf16(afr[tm].v, bfr[tn].v, acc[tm][tn]);

        __syncthreads();
    }

    // ---- epilogue ----
    #pragma unroll
    for (int tm = 0; tm < 4; ++tm) {
        #pragma unroll
        for (int tn = 0; tn < 2; ++tn) {
            #pragma unroll
            for (int r = 0; r < 8; ++r) {
                int m = m0 + wm * 64 + tm * 16 + lhalf * 8 + r;
                int n = n0 + wn * 32 + tn * 16 + lmod;
                if (m < M) {
                    float v = acc[tm][tn][r];
                    if (FLAGS & 4)  v += bias[n];
                    if (FLAGS & 8)  v = fmaxf(v, 0.f);
                    if (FLAGS & 16) v += resid[(size_t)m * N + n];
                    size_t o = (size_t)m * N + n;
                    if (FLAGS & 2) outf[o] = v;
                    if (FLAGS & 1) outb[o] = (__bf16)v;
                }
            }
        }
    }
}

// ---------------------------------------------------------------------------
// Flash attention: one wave per (b, h, 16-query tile).
// Q,K,V,Ctx are bf16 [B, S, H*HD]; HD=64; scale = 1/8.
// ---------------------------------------------------------------------------
__global__ __launch_bounds__(32)
void xf_attn(const __bf16* __restrict__ Q, const __bf16* __restrict__ Kmat,
             const __bf16* __restrict__ V, __bf16* __restrict__ Ctx) {
    const int lane  = threadIdx.x;
    const int lmod  = lane & 15;
    const int lhalf = lane >> 4;
    const int qt    = blockIdx.x;
    const int bh    = blockIdx.y;
    const int b     = bh / HH;
    const int h     = bh % HH;
    const int qbase = qt * 16;
    const int KT    = HH * HDIM;   // 1024

    // Q fragments (A-matrix 16x32, two hd chunks)
    FragB qf[2];
    {
        int qrow = qbase + lmod; if (qrow >= SS) qrow = SS - 1;
        const __bf16* qp = Q + ((size_t)(b * SS + qrow)) * KT + h * HDIM;
        #pragma unroll
        for (int c = 0; c < 2; ++c) {
            qf[c].q[0] = *(const uint4*)(qp + c * 32 + lhalf * 16);
            qf[c].q[1] = *(const uint4*)(qp + c * 32 + lhalf * 16 + 8);
        }
    }

    f32x8 o[4];
    #pragma unroll
    for (int j = 0; j < 4; ++j) o[j] = (f32x8){0.f,0.f,0.f,0.f,0.f,0.f,0.f,0.f};
    float mrun[8], lrun[8];
    #pragma unroll
    for (int r = 0; r < 8; ++r) { mrun[r] = -3.0e38f; lrun[r] = 0.f; }

    __shared__ __bf16 psh[16 * 32];

    for (int kb = 0; kb < SS; kb += 32) {
        // ---- scores: two 16x16 tiles over this 32-key chunk ----
        f32x8 s[2];
        #pragma unroll
        for (int tk = 0; tk < 2; ++tk) {
            int key = kb + tk * 16 + lmod; if (key >= SS) key = SS - 1;
            const __bf16* kp = Kmat + ((size_t)(b * SS + key)) * KT + h * HDIM;
            FragB kf0, kf1;
            kf0.q[0] = *(const uint4*)(kp + lhalf * 16);
            kf0.q[1] = *(const uint4*)(kp + lhalf * 16 + 8);
            kf1.q[0] = *(const uint4*)(kp + 32 + lhalf * 16);
            kf1.q[1] = *(const uint4*)(kp + 32 + lhalf * 16 + 8);
            f32x8 z = (f32x8){0.f,0.f,0.f,0.f,0.f,0.f,0.f,0.f};
            z = wmma_bf16(qf[0].v, kf0.v, z);
            z = wmma_bf16(qf[1].v, kf1.v, z);
            s[tk] = z;
        }
        // scale + key mask
        #pragma unroll
        for (int tk = 0; tk < 2; ++tk) {
            int keyn = kb + tk * 16 + lmod;
            #pragma unroll
            for (int r = 0; r < 8; ++r) {
                float v = s[tk][r] * 0.125f;
                if (keyn >= SS) v = -3.0e38f;
                s[tk][r] = v;
            }
        }
        // ---- online softmax update (rows = r + 8*lhalf, cols = lanes) ----
        #pragma unroll
        for (int r = 0; r < 8; ++r) {
            float rmax = fmaxf(s[0][r], s[1][r]);
            #pragma unroll
            for (int off = 1; off < 16; off <<= 1)
                rmax = fmaxf(rmax, __shfl_xor(rmax, off, 32));
            float newm  = fmaxf(mrun[r], rmax);
            float alpha = __expf(mrun[r] - newm);
            float p0 = __expf(s[0][r] - newm);
            float p1 = __expf(s[1][r] - newm);
            s[0][r] = p0; s[1][r] = p1;
            float part = p0 + p1;
            #pragma unroll
            for (int off = 1; off < 16; off <<= 1)
                part += __shfl_xor(part, off, 32);
            lrun[r] = lrun[r] * alpha + part;
            mrun[r] = newm;
            #pragma unroll
            for (int j = 0; j < 4; ++j) o[j][r] *= alpha;
        }
        // ---- P: C-layout -> A-fragment via wave-private LDS ----
        #pragma unroll
        for (int tk = 0; tk < 2; ++tk)
            #pragma unroll
            for (int r = 0; r < 8; ++r)
                psh[(r + 8 * lhalf) * 32 + tk * 16 + lmod] = (__bf16)s[tk][r];
        asm volatile("s_wait_dscnt 0" ::: "memory");
        __syncthreads();
        FragB pf;
        {
            const __bf16* pp = psh + lmod * 32 + lhalf * 16;
            pf.q[0] = *(const uint4*)(pp);
            pf.q[1] = *(const uint4*)(pp + 8);
        }
        asm volatile("" ::: "memory");
        // ---- O += P * V  (4 B-fragments over hd) ----
        #pragma unroll
        for (int j = 0; j < 4; ++j) {
            FragB vf;
            int hd = j * 16 + lmod;
            #pragma unroll
            for (int tt = 0; tt < 16; ++tt) {
                int key = kb + lhalf * 16 + tt; if (key >= SS) key = SS - 1;
                vf.e[tt] = V[((size_t)(b * SS + key)) * KT + h * HDIM + hd];
            }
            o[j] = wmma_bf16(pf.v, vf.v, o[j]);
        }
        __syncthreads();
    }

    // ---- normalize & store ctx ----
    #pragma unroll
    for (int j = 0; j < 4; ++j) {
        #pragma unroll
        for (int r = 0; r < 8; ++r) {
            int m = r + 8 * lhalf;
            int qrow = qbase + m;
            if (qrow < SS) {
                float v = o[j][r] / lrun[r];
                Ctx[((size_t)(b * SS + qrow)) * KT + h * HDIM + j * 16 + lmod] = (__bf16)v;
            }
        }
    }
}

// ---------------------------------------------------------------------------
// LayerNorm over (S,D) per sample: reduce then apply (optionally bf16 copy)
// ---------------------------------------------------------------------------
__global__ __launch_bounds__(256)
void xf_ln_reduce(const float* __restrict__ x, float* __restrict__ stats, int per) {
    const int b = blockIdx.x;
    const float* p = x + (size_t)b * per;
    float s = 0.f, sq = 0.f;
    for (int i = threadIdx.x; i < per; i += 256) { float v = p[i]; s += v; sq += v * v; }
    __shared__ float sh[512];
    sh[threadIdx.x] = s; sh[256 + threadIdx.x] = sq;
    __syncthreads();
    for (int st = 128; st > 0; st >>= 1) {
        if ((int)threadIdx.x < st) {
            sh[threadIdx.x]       += sh[threadIdx.x + st];
            sh[256 + threadIdx.x] += sh[256 + threadIdx.x + st];
        }
        __syncthreads();
    }
    if (threadIdx.x == 0) {
        float mean = sh[0] / (float)per;
        float var  = sh[256] / (float)per - mean * mean;
        stats[b * 2]     = mean;
        stats[b * 2 + 1] = rsqrtf(var + 1e-5f);
    }
}

__global__ void xf_ln_apply(const float* __restrict__ x, float* __restrict__ yf,
                            __bf16* __restrict__ yb, const float* __restrict__ stats, int per) {
    int b = blockIdx.y;
    int i = blockIdx.x * blockDim.x + threadIdx.x;
    if (i >= per) return;
    float mean = stats[b * 2], rstd = stats[b * 2 + 1];
    size_t o = (size_t)b * per + i;
    float v = (x[o] - mean) * rstd;
    yf[o] = v;
    if (yb) yb[o] = (__bf16)v;
}

// ---------------------------------------------------------------------------
// mean over S, then small classifier GEMM (fp32, memory-trivial)
// ---------------------------------------------------------------------------
__global__ void xf_pool(const float* __restrict__ ffn, float* __restrict__ pooled) {
    int idx = blockIdx.x * blockDim.x + threadIdx.x;
    if (idx >= BB * DD) return;
    int b = idx / DD, d = idx % DD;
    float s = 0.f;
    for (int t = 0; t < SS; ++t) s += ffn[((size_t)b * SS + t) * DD + d];
    pooled[idx] = s * (1.0f / (float)SS);
}

__global__ void xf_outproj(const float* __restrict__ pooled, const float* __restrict__ Wout,
                           float* __restrict__ out) {
    int idx = blockIdx.x * blockDim.x + threadIdx.x;
    if (idx >= BB * OUTN) return;
    int b = idx / OUTN, oc = idx % OUTN;
    float s = 0.f;
    for (int d = 0; d < DD; ++d) s += pooled[b * DD + d] * Wout[(size_t)d * OUTN + oc];
    out[idx] = s;
}

// ---------------------------------------------------------------------------
// Host side
// ---------------------------------------------------------------------------
template <int FLAGS>
static void launch_gemm(const __bf16* A, const __bf16* Bw, int M, int N, int Kd,
                        const float* bias, const float* resid,
                        __bf16* ob, float* of, hipStream_t s) {
    dim3 g((N + 127) / 128, (M + 127) / 128), blk(256);
    xf_gemm<FLAGS><<<g, blk, 0, s>>>(A, Bw, M, N, Kd, bias, resid, ob, of);
}

extern "C" void kernel_launch(void* const* d_in, const int* in_sizes, int n_in,
                              void* d_out, int out_size, void* d_ws, size_t ws_size,
                              hipStream_t stream) {
    (void)in_sizes; (void)n_in; (void)out_size; (void)ws_size;
    const float* x    = (const float*)d_in[0];
    const float* Wq   = (const float*)d_in[1];
    const float* Wk   = (const float*)d_in[2];
    const float* Wv   = (const float*)d_in[3];
    const float* Wo   = (const float*)d_in[4];
    const float* W1   = (const float*)d_in[5];
    const float* b1   = (const float*)d_in[6];
    const float* W2   = (const float*)d_in[7];
    const float* b2   = (const float*)d_in[8];
    const float* Wout = (const float*)d_in[9];
    float* out = (float*)d_out;

    char* ws = (char*)d_ws;
    size_t off = 0;
    auto alloc = [&](size_t bytes) -> void* {
        void* p = (void*)(ws + off);
        off += (bytes + 255) & ~(size_t)255;
        return p;
    };

    const size_t nBSD = (size_t)BB * SS * DD;   // 8,192,000
    const size_t nBSM = (size_t)BB * SS * MIDD; // 32,768,000

    __bf16* wq_bf = (__bf16*)alloc((size_t)DD * KK * 2);
    __bf16* wk_bf = (__bf16*)alloc((size_t)DD * KK * 2);
    __bf16* wv_bf = (__bf16*)alloc((size_t)DD * KK * 2);
    __bf16* wo_bf = (__bf16*)alloc((size_t)KK * DD * 2);
    __bf16* w1_bf = (__bf16*)alloc((size_t)DD * MIDD * 2);
    __bf16* w2_bf = (__bf16*)alloc((size_t)MIDD * DD * 2);
    __bf16* xpe_bf = (__bf16*)alloc(nBSD * 2);        // later reused as ctx
    __bf16* q_bf  = (__bf16*)alloc(nBSD * 2);
    __bf16* k_bf  = (__bf16*)alloc(nBSD * 2);
    __bf16* v_bf  = (__bf16*)alloc(nBSD * 2);
    __bf16* ao_bf = (__bf16*)alloc(nBSD * 2);         // LN'd attn out, bf16
    __bf16* h_bf  = (__bf16*)alloc(nBSM * 2);
    float*  attn_f = (float*)alloc(nBSD * 4);         // ctx@Wo, LN'd in place
    float*  ffn_f  = (float*)alloc(nBSD * 4);         // FFN out, LN'd in place
    float*  stats  = (float*)alloc(2 * BB * 4);
    float*  pooled = (float*)alloc((size_t)BB * DD * 4);

    const int M = BB * SS;   // 8000
    dim3 b256(256);
    auto cgrid = [](size_t n) { return dim3((unsigned)((n + 255) / 256)); };

    // weights -> bf16
    xf_f2bf<<<cgrid((size_t)DD*KK),   b256, 0, stream>>>(Wq, wq_bf, (size_t)DD*KK);
    xf_f2bf<<<cgrid((size_t)DD*KK),   b256, 0, stream>>>(Wk, wk_bf, (size_t)DD*KK);
    xf_f2bf<<<cgrid((size_t)DD*KK),   b256, 0, stream>>>(Wv, wv_bf, (size_t)DD*KK);
    xf_f2bf<<<cgrid((size_t)KK*DD),   b256, 0, stream>>>(Wo, wo_bf, (size_t)KK*DD);
    xf_f2bf<<<cgrid((size_t)DD*MIDD), b256, 0, stream>>>(W1, w1_bf, (size_t)DD*MIDD);
    xf_f2bf<<<cgrid((size_t)MIDD*DD), b256, 0, stream>>>(W2, w2_bf, (size_t)MIDD*DD);

    // x + posenc -> bf16
    xf_posenc<<<cgrid(nBSD), b256, 0, stream>>>(x, xpe_bf);

    // QKV projections (WMMA)
    launch_gemm<1>(xpe_bf, wq_bf, M, KK, DD, nullptr, nullptr, q_bf, nullptr, stream);
    launch_gemm<1>(xpe_bf, wk_bf, M, KK, DD, nullptr, nullptr, k_bf, nullptr, stream);
    launch_gemm<1>(xpe_bf, wv_bf, M, KK, DD, nullptr, nullptr, v_bf, nullptr, stream);

    // flash attention -> ctx (reuse xpe buffer)
    __bf16* ctx_bf = xpe_bf;
    dim3 agrid((SS + 15) / 16, BB * HH);
    xf_attn<<<agrid, dim3(32), 0, stream>>>(q_bf, k_bf, v_bf, ctx_bf);

    // ctx @ Wo -> f32, LayerNorm over (S,D) in place (+ bf16 copy)
    launch_gemm<2>(ctx_bf, wo_bf, M, DD, KK, nullptr, nullptr, nullptr, attn_f, stream);
    xf_ln_reduce<<<dim3(BB), b256, 0, stream>>>(attn_f, stats, SS * DD);
    xf_ln_apply<<<dim3((SS*DD + 255)/256, BB), b256, 0, stream>>>(attn_f, attn_f, ao_bf, stats, SS * DD);

    // FFN: relu(ao @ W1 + b1) -> h ;  h @ W2 + b2 + ao -> f ; LN in place
    launch_gemm<1|4|8>(ao_bf, w1_bf, M, MIDD, DD, b1, nullptr, h_bf, nullptr, stream);
    launch_gemm<2|4|16>(h_bf, w2_bf, M, DD, MIDD, b2, attn_f, nullptr, ffn_f, stream);
    xf_ln_reduce<<<dim3(BB), b256, 0, stream>>>(ffn_f, stats, SS * DD);
    xf_ln_apply<<<dim3((SS*DD + 255)/256, BB), b256, 0, stream>>>(ffn_f, ffn_f, nullptr, stats, SS * DD);

    // pool + classifier
    xf_pool<<<cgrid((size_t)BB * DD), b256, 0, stream>>>(ffn_f, pooled);
    xf_outproj<<<cgrid((size_t)BB * OUTN), b256, 0, stream>>>(pooled, Wout, out);
}
